// AtencaoSelfMultiCabeca_34600256537391
// MI455X (gfx1250) — matrix operand order repaired and checked by reading, not compiled, not run on hardware
//
#include <hip/hip_runtime.h>
#include <hip/hip_bf16.h>

// ---------------------------------------------------------------------------
// Types for CDNA5 WMMA (wave32)
// ---------------------------------------------------------------------------
typedef __attribute__((ext_vector_type(16))) __bf16     v16bf;
typedef __attribute__((ext_vector_type(8)))  float      v8f;
typedef __attribute__((ext_vector_type(4)))  float      f32x4;
typedef __attribute__((ext_vector_type(4)))  unsigned   u32x4;

union ABf {            // one 16x32 bf16 WMMA operand fragment (8 VGPRs/lane)
    v16bf v;
    u32x4 q[2];
    unsigned u[8];
};

__device__ __forceinline__ unsigned short f2bf(float f) {
    union { float f; unsigned u; } a; a.f = f;
    unsigned r = a.u + 0x7FFFu + ((a.u >> 16) & 1u);   // RNE
    return (unsigned short)(r >> 16);
}
__device__ __forceinline__ unsigned pack2(float lo, float hi) {
    return (unsigned)f2bf(lo) | ((unsigned)f2bf(hi) << 16);
}
__device__ __forceinline__ v8f vzero8() {
    v8f z;
#pragma unroll
    for (int i = 0; i < 8; ++i) z[i] = 0.0f;
    return z;
}
__device__ __forceinline__ v8f wmma_bf16(const v16bf& a, const v16bf& b, const v8f& c) {
    return __builtin_amdgcn_wmma_f32_16x16x32_bf16(
        false, a, false, b, (short)0, c, false, false);
}

// low 32 bits of a flat LDS pointer == byte offset into the wave's LDS space
__device__ __forceinline__ unsigned lds_off(const void* p) {
    return (unsigned)(uintptr_t)p;
}
// CDNA5 async DMA: global -> LDS, 16B per lane, tracked by ASYNCcnt
__device__ __forceinline__ void async_ld_b128(unsigned lds_byte_off, const void* gptr) {
    asm volatile("global_load_async_to_lds_b128 %0, %1, off"
                 :: "v"(lds_byte_off), "v"((unsigned long long)(uintptr_t)gptr)
                 : "memory");
}
__device__ __forceinline__ void wait_async0() {
    asm volatile("s_wait_asynccnt 0x0" ::: "memory");
}

// ---------------------------------------------------------------------------
// Problem constants (b=2, n=1024, d_m=2048, heads=16, d_k=128)
// ---------------------------------------------------------------------------
#define BB   2
#define NN   1024
#define DM   2048
#define NH   16
#define DK   128
#define QKVN (3 * DM)          // 6144

// LDS row pitch (bf16 elems): 40 -> 80 B rows (16B-aligned b128 slots)
#define LP    40
#define TILEB (128 * LP)       // ushorts per 128-row LDS tile (10240 B)
#define BTILE (256 * LP)       // ushorts per 256-row LDS tile (20480 B)

// ---------------------------------------------------------------------------
// Kernel 0a: f32 -> bf16 bulk convert (8 elems/thread)
// ---------------------------------------------------------------------------
__global__ __launch_bounds__(256) void cvt_bf16(
    const float* __restrict__ src, unsigned short* __restrict__ dst)
{
    const size_t i = ((size_t)blockIdx.x * 256 + threadIdx.x) * 8;
    f32x4 a = *reinterpret_cast<const f32x4*>(&src[i]);
    f32x4 b = *reinterpret_cast<const f32x4*>(&src[i + 4]);
    u32x4 o;
    o[0] = pack2(a[0], a[1]); o[1] = pack2(a[2], a[3]);
    o[2] = pack2(b[0], b[1]); o[3] = pack2(b[2], b[3]);
    *reinterpret_cast<u32x4*>(&dst[i]) = o;
}

// ---------------------------------------------------------------------------
// Kernel 0b: W[K][N] f32  ->  Wt[N][K] bf16  (32x32 LDS tiles)
// ---------------------------------------------------------------------------
__global__ __launch_bounds__(256) void transpose_cvt(
    const float* __restrict__ W, unsigned short* __restrict__ Wt, int K, int N)
{
    __shared__ float tile[32][33];
    const int tx = threadIdx.x & 31, ty = threadIdx.x >> 5;      // 32 x 8
    const int k0 = blockIdx.x * 32, n0 = blockIdx.y * 32;
#pragma unroll
    for (int i = 0; i < 4; ++i)
        tile[ty + i * 8][tx] = W[(size_t)(k0 + ty + i * 8) * N + n0 + tx];
    __syncthreads();
#pragma unroll
    for (int i = 0; i < 4; ++i)
        Wt[(size_t)(n0 + ty + i * 8) * K + k0 + tx] = f2bf(tile[tx][ty + i * 8]);
}

// ---------------------------------------------------------------------------
// Kernels 1/4 & 4/4: bf16 WMMA GEMM, C[M,N] = A[M,K] @ Bt[N,K]^T (+bias)
//   Block tile 128x256, K-step 32, 8 waves as 2x4, wave tile 64x64
//   -> 16 WMMA per 16 ds_load_b128 per wave per K-step.
//   Double-buffered LDS fed by global_load_async_to_lds_b128 (ASYNCcnt).
// ---------------------------------------------------------------------------
template<bool HAS_BIAS>
__global__ __launch_bounds__(256) void gemm_wmma_bf16(
    const unsigned short* __restrict__ A,   // [M][K] bf16
    const unsigned short* __restrict__ Bt,  // [N][K] bf16
    const float* __restrict__ bias, float* __restrict__ C,
    int M, int N, int K)
{
    __shared__ __align__(16) unsigned short As[2 * TILEB];   // 20.5 KB
    __shared__ __align__(16) unsigned short Bs[2 * BTILE];   // 41.0 KB

    const int tid  = threadIdx.x;
    const int lane = tid & 31;
    const int w    = tid >> 5;
    const int hl   = lane >> 4;
    const int ln   = lane & 15;
    const int wm   = w >> 2;          // 0..1 : 64-row strip
    const int wn   = w & 3;           // 0..3 : 64-col strip
    const size_t m0 = (size_t)blockIdx.x * 128;
    const size_t n0 = (size_t)blockIdx.y * 256;

    const unsigned asBase = lds_off(As);
    const unsigned bsBase = lds_off(Bs);

    v8f acc[4][4];
#pragma unroll
    for (int mt = 0; mt < 4; ++mt)
#pragma unroll
        for (int nt = 0; nt < 4; ++nt) acc[mt][nt] = vzero8();

    // stage A(128x32) + B(256x32) bf16 tiles into LDS buffer `buf`
    auto stage = [&](int kt, int buf) {
        const int k0 = kt << 5;
#pragma unroll
        for (int r = 0; r < 2; ++r) {          // A: 512 x 16B segments
            const int e   = r * 256 + tid;
            const int row = e >> 2;            // 0..127
            const int kq  = (e & 3) * 8;       // 0,8,16,24
            async_ld_b128(asBase + (unsigned)(buf * TILEB + row * LP + kq) * 2,
                          &A[(m0 + row) * (size_t)K + k0 + kq]);
        }
#pragma unroll
        for (int r = 0; r < 4; ++r) {          // B: 1024 x 16B segments
            const int e   = r * 256 + tid;
            const int row = e >> 2;            // 0..255
            const int kq  = (e & 3) * 8;
            async_ld_b128(bsBase + (unsigned)(buf * BTILE + row * LP + kq) * 2,
                          &Bt[(n0 + row) * (size_t)K + k0 + kq]);
        }
    };

    const int ksteps = K >> 5;
    stage(0, 0);
    int buf = 0;

    for (int kt = 0; kt < ksteps; ++kt) {
        wait_async0();          // my async copies into `buf` have landed
        __syncthreads();        // everyone's have; prior ds reads drained

        if (kt + 1 < ksteps) stage(kt + 1, buf ^ 1);

        const unsigned short* as = &As[buf * TILEB];
        const unsigned short* bs = &Bs[buf * BTILE];

        ABf afr[4];
#pragma unroll
        for (int mt = 0; mt < 4; ++mt) {
            const int base = (wm * 64 + mt * 16 + ln) * LP + hl * 8;
            afr[mt].q[0] = *reinterpret_cast<const u32x4*>(&as[base]);
            afr[mt].q[1] = *reinterpret_cast<const u32x4*>(&as[base + 16]);
        }
#pragma unroll
        for (int nt = 0; nt < 4; ++nt) {
            ABf bfr;
            const int base = (wn * 64 + nt * 16 + ln) * LP + hl * 8;
            bfr.q[0] = *reinterpret_cast<const u32x4*>(&bs[base]);
            bfr.q[1] = *reinterpret_cast<const u32x4*>(&bs[base + 16]);
#pragma unroll
            for (int mt = 0; mt < 4; ++mt)
                acc[mt][nt] = wmma_bf16(afr[mt].v, bfr.v, acc[mt][nt]);
        }
        buf ^= 1;
    }

    // epilogue: C/D layout -> lane = col, vgpr r -> row r+8*hl
#pragma unroll
    for (int mt = 0; mt < 4; ++mt)
#pragma unroll
        for (int nt = 0; nt < 4; ++nt) {
            const size_t gcol = n0 + wn * 64 + nt * 16 + ln;
            const float  bv   = HAS_BIAS ? bias[gcol] : 0.0f;
#pragma unroll
            for (int r = 0; r < 8; ++r) {
                const size_t grow = m0 + wm * 64 + mt * 16 + r + 8 * hl;
                C[grow * (size_t)N + gcol] = acc[mt][nt][r] + bv;
            }
        }
}

// ---------------------------------------------------------------------------
// Kernel 2/4: RoPE on Q,K + bf16 repack.  Q pre-scaled by 1/sqrt(d_k).
//   Q,K -> [b][h][n][128]  (row-major per head)
//   V   -> [b][h][128][n]  (pre-transposed so P@V B-fragments are contiguous)
// ---------------------------------------------------------------------------
__global__ __launch_bounds__(256) void rope_cvt(
    const float* __restrict__ qkv,
    unsigned short* __restrict__ Qb,
    unsigned short* __restrict__ Kb,
    unsigned short* __restrict__ Vt)
{
    const unsigned idx = blockIdx.x * 256u + threadIdx.x;   // b*n*h*64 threads
    const int j = idx & 63;
    const int h = (idx >> 6) & 15;
    const int n = (idx >> 10) & 1023;
    const int b = idx >> 20;
    const int bh = b * NH + h;

    const size_t rb = ((size_t)(b * NN + n)) * QKVN + (size_t)h * DK;
    const float q1 = qkv[rb + j],            q2 = qkv[rb + j + 64];
    const float k1 = qkv[rb + DM + j],       k2 = qkv[rb + DM + j + 64];
    const float v1 = qkv[rb + 2 * DM + j],   v2 = qkv[rb + 2 * DM + j + 64];

    const float inv = __powf(10000.0f, -(float)j * (1.0f / 64.0f));
    float s, c;
    __sincosf((float)n * inv, &s, &c);

    const float qs = 0.08838834764831845f;                  // 1/sqrt(128)
    const size_t db = (((size_t)bh) * NN + n) * DK;
    Qb[db + j]      = f2bf((q1 * c - q2 * s) * qs);
    Qb[db + j + 64] = f2bf((q2 * c + q1 * s) * qs);
    Kb[db + j]      = f2bf(k1 * c - k2 * s);
    Kb[db + j + 64] = f2bf(k2 * c + k1 * s);

    const size_t vb = (size_t)bh * DK * NN;                 // [dk][n]
    Vt[vb + (size_t)(j)      * NN + n] = f2bf(v1);
    Vt[vb + (size_t)(j + 64) * NN + n] = f2bf(v2);
}

// ---------------------------------------------------------------------------
// Kernel 3/4: causal flash attention (barrier-free waves).
//   Exactly ONE 32-key chunk per wave straddles the diagonal -> masked tail;
//   all earlier chunks take the unmasked fast path (no v_cmp/v_cndmask).
// ---------------------------------------------------------------------------
template<bool MASKED>
__device__ __forceinline__ void attn_chunk(
    int kc, int qbase, int hl, int ln, int wbase,
    const unsigned short* __restrict__ kp,
    const unsigned short* __restrict__ vp,
    unsigned short* __restrict__ Pl,
    const ABf (&qf)[4], v8f (&O)[8], float (&mrow)[8], float (&lrow)[8])
{
    // ---- S = Q K^T for two 16-key tiles (Q pre-scaled) -------------------
    v8f S[2];
#pragma unroll
    for (int nt = 0; nt < 2; ++nt) {
        S[nt] = vzero8();
        const size_t keyrow = (size_t)(kc * 32 + nt * 16 + ln) * DK;
#pragma unroll
        for (int f = 0; f < 4; ++f) {
            ABf kf;
            const size_t base = keyrow + f * 32 + hl * 8;
            kf.q[0] = *reinterpret_cast<const u32x4*>(&kp[base]);
            kf.q[1] = *reinterpret_cast<const u32x4*>(&kp[base + 16]);
            S[nt] = wmma_bf16(qf[f].v, kf.v, S[nt]);
        }
    }

    // ---- (mask) + row max ------------------------------------------------
    float chmax[8];
#pragma unroll
    for (int r = 0; r < 8; ++r) chmax[r] = -1e30f;
#pragma unroll
    for (int nt = 0; nt < 2; ++nt) {
        const int key = kc * 32 + nt * 16 + ln;
#pragma unroll
        for (int r = 0; r < 8; ++r) {
            float sv = S[nt][r];
            if (MASKED) {
                const int row = qbase + r + 8 * hl;
                sv = (key <= row) ? sv : -1e30f;
                S[nt][r] = sv;
            }
            chmax[r] = fmaxf(chmax[r], sv);
        }
    }
#pragma unroll
    for (int r = 0; r < 8; ++r) {
        chmax[r] = fmaxf(chmax[r], __shfl_xor(chmax[r], 1, 32));
        chmax[r] = fmaxf(chmax[r], __shfl_xor(chmax[r], 2, 32));
        chmax[r] = fmaxf(chmax[r], __shfl_xor(chmax[r], 4, 32));
        chmax[r] = fmaxf(chmax[r], __shfl_xor(chmax[r], 8, 32));
    }

    // ---- online softmax update -------------------------------------------
    float alpha[8];
#pragma unroll
    for (int r = 0; r < 8; ++r) {
        const float mnew = fmaxf(mrow[r], chmax[r]);
        alpha[r] = __expf(mrow[r] - mnew);
        mrow[r]  = mnew;
        lrow[r] *= alpha[r];
    }
#pragma unroll
    for (int nt = 0; nt < 8; ++nt)
#pragma unroll
        for (int r = 0; r < 8; ++r) O[nt][r] *= alpha[r];

    // ---- P = exp(S-m): row sums + bf16 stash into per-wave LDS pad -------
    float psum[8];
#pragma unroll
    for (int r = 0; r < 8; ++r) psum[r] = 0.0f;
#pragma unroll
    for (int nt = 0; nt < 2; ++nt)
#pragma unroll
        for (int r = 0; r < 8; ++r) {
            const float pv = __expf(S[nt][r] - mrow[r]);
            psum[r] += pv;
            Pl[wbase + (r + 8 * hl) * LP + nt * 16 + ln] = f2bf(pv);
        }
#pragma unroll
    for (int r = 0; r < 8; ++r) {
        psum[r] += __shfl_xor(psum[r], 1, 32);
        psum[r] += __shfl_xor(psum[r], 2, 32);
        psum[r] += __shfl_xor(psum[r], 4, 32);
        psum[r] += __shfl_xor(psum[r], 8, 32);
        lrow[r] += psum[r];
    }

    // wave-local LDS transpose RAW: drain DS pipe (CDNA5 split counter)
    asm volatile("s_wait_dscnt 0x0" ::: "memory");

    ABf pf;
    {
        const int base = wbase + ln * LP + hl * 8;
        pf.q[0] = *reinterpret_cast<const u32x4*>(&Pl[base]);
        pf.q[1] = *reinterpret_cast<const u32x4*>(&Pl[base + 16]);
    }

    // ---- O += P @ V : V fragments direct from transposed global ----------
#pragma unroll
    for (int nt = 0; nt < 8; ++nt) {
        ABf vf;
        const size_t base = (size_t)(nt * 16 + ln) * NN + kc * 32 + hl * 8;
        vf.q[0] = *reinterpret_cast<const u32x4*>(&vp[base]);
        vf.q[1] = *reinterpret_cast<const u32x4*>(&vp[base + 16]);
        O[nt] = wmma_bf16(pf.v, vf.v, O[nt]);
    }
}

__global__ __launch_bounds__(256) void flash_attn(
    const unsigned short* __restrict__ Qb,
    const unsigned short* __restrict__ Kb,
    const unsigned short* __restrict__ Vt,
    unsigned short* __restrict__ Out)   // [b*n][2048] token-major bf16
{
    __shared__ __align__(16) unsigned short Pl[8 * 16 * LP];   // per-wave pad

    const int tid  = threadIdx.x;
    const int lane = tid & 31;
    const int w    = tid >> 5;
    const int hl   = lane >> 4;
    const int ln   = lane & 15;

    const int bh = blockIdx.x;                 // b*16 + h
    const int b  = bh >> 4, h = bh & 15;
    const int qbase = blockIdx.y * 128 + w * 16;
    const int wbase = w * 16 * LP;

    const unsigned short* qp = Qb + (size_t)bh * NN * DK;
    const unsigned short* kp = Kb + (size_t)bh * NN * DK;
    const unsigned short* vp = Vt + (size_t)bh * DK * NN;    // [dk][n]

    // Q fragments resident in registers: 4 x (16x32) bf16
    ABf qf[4];
    {
        const size_t qrow = (size_t)(qbase + ln) * DK;
#pragma unroll
        for (int f = 0; f < 4; ++f) {
            const size_t base = qrow + f * 32 + hl * 8;
            qf[f].q[0] = *reinterpret_cast<const u32x4*>(&qp[base]);
            qf[f].q[1] = *reinterpret_cast<const u32x4*>(&qp[base + 16]);
        }
    }

    v8f O[8];
#pragma unroll
    for (int nt = 0; nt < 8; ++nt) O[nt] = vzero8();
    float mrow[8], lrow[8];
#pragma unroll
    for (int r = 0; r < 8; ++r) { mrow[r] = -1e30f; lrow[r] = 0.0f; }

    // chunks strictly below the diagonal: unmasked fast path
    const int nfull = (qbase + 1) >> 5;
    for (int kc = 0; kc < nfull; ++kc)
        attn_chunk<false>(kc, qbase, hl, ln, wbase, kp, vp, Pl, qf, O, mrow, lrow);
    // the single diagonal-straddling chunk: masked
    attn_chunk<true>(nfull, qbase, hl, ln, wbase, kp, vp, Pl, qf, O, mrow, lrow);

    // ---- normalize (1 rcp per row) and write [token][h*128+dk] bf16 ------
    float rinv[8];
#pragma unroll
    for (int r = 0; r < 8; ++r) rinv[r] = 1.0f / lrow[r];
#pragma unroll
    for (int nt = 0; nt < 8; ++nt)
#pragma unroll
        for (int r = 0; r < 8; ++r) {
            const int row = qbase + r + 8 * hl;
            Out[((size_t)(b * NN + row)) * DM + h * DK + nt * 16 + ln] =
                f2bf(O[nt][r] * rinv[r]);
        }
}

// ---------------------------------------------------------------------------
// Host launcher
// ---------------------------------------------------------------------------
extern "C" void kernel_launch(void* const* d_in, const int* in_sizes, int n_in,
                              void* d_out, int out_size, void* d_ws, size_t ws_size,
                              hipStream_t stream)
{
    const float* x    = (const float*)d_in[0];   // [2,1024,2048]
    // d_in[1] = causal mask (tril) -- implicit in the attention kernel
    const float* Wqkv = (const float*)d_in[2];   // [2048,6144]
    const float* Wout = (const float*)d_in[3];   // [2048,2048]
    const float* bout = (const float*)d_in[4];   // [2048]
    float* out = (float*)d_out;                  // [2,1024,2048]

    char* ws = (char*)d_ws;
    const size_t QKV_B  = (size_t)BB * NN * QKVN * 4;   // 50.3 MB  f32 qkv
    const size_t HDB    = (size_t)BB * NH * NN * DK * 2;//  8.4 MB  bf16/head buf
    const size_t XB_B   = (size_t)BB * NN * DM * 2;     //  8.4 MB  bf16 x
    const size_t WQKV_B = (size_t)DM * QKVN * 2;        // 25.2 MB  bf16 WqkvT
    size_t off = 0;
    float*          qkv   = (float*)(ws + off);          off += QKV_B;
    unsigned short* Qb    = (unsigned short*)(ws + off); off += HDB;
    unsigned short* Kb    = (unsigned short*)(ws + off); off += HDB;
    unsigned short* Vt    = (unsigned short*)(ws + off); off += HDB;
    unsigned short* attn  = (unsigned short*)(ws + off); off += HDB;
    unsigned short* xb    = (unsigned short*)(ws + off); off += XB_B;
    unsigned short* WqkvT = (unsigned short*)(ws + off); off += WQKV_B;
    unsigned short* WoutT = (unsigned short*)(ws + off);

    // 0) one-time precision/layout prep (bandwidth-trivial vs compute)
    cvt_bf16<<<(BB * NN * DM) / (256 * 8), 256, 0, stream>>>(x, xb);
    transpose_cvt<<<dim3(DM / 32, QKVN / 32), 256, 0, stream>>>(Wqkv, WqkvT, DM, QKVN);
    transpose_cvt<<<dim3(DM / 32, DM / 32),   256, 0, stream>>>(Wout, WoutT, DM, DM);

    // 1) QKV projection: [2048,6144] = xb @ WqkvT^T
    gemm_wmma_bf16<false><<<dim3(BB * NN / 128, QKVN / 256), 256, 0, stream>>>(
        xb, WqkvT, nullptr, qkv, BB * NN, QKVN, DM);

    // 2) RoPE + bf16 repack (Q pre-scaled, V pre-transposed per head)
    rope_cvt<<<(BB * NN * NH * 64) / 256, 256, 0, stream>>>(qkv, Qb, Kb, Vt);

    // 3) causal flash attention -> token-major bf16 [2048, 2048]
    flash_attn<<<dim3(BB * NH, NN / 128), 256, 0, stream>>>(Qb, Kb, Vt, attn);

    // 4) output projection + bias -> f32 d_out
    gemm_wmma_bf16<true><<<dim3(BB * NN / 128, DM / 256), 256, 0, stream>>>(
        attn, WoutT, bout, out, BB * NN, DM, DM);
}
